// MaskedCrossAttention_48962627174864
// MI455X (gfx1250) — matrix operand order repaired and checked by reading, not compiled, not run on hardware
//
#include <hip/hip_runtime.h>
#include <hip/hip_bf16.h>
#include <math.h>

// ---------- types ----------
typedef __attribute__((ext_vector_type(16))) _Float16 v16h;
typedef __attribute__((ext_vector_type(8)))  _Float16 v8h;
typedef __attribute__((ext_vector_type(4)))  _Float16 v4h;
typedef __attribute__((ext_vector_type(8)))  float    v8f;
typedef __attribute__((ext_vector_type(4)))  float    v4f;

#define DIMC      1024
#define INNER     512
#define NQ        2048   // query length per batch
#define NJ        1024   // key/value length per batch
#define NB        2
#define NH        8
#define DHEAD     64
#define SCALE_Q   0.125f // 64^-0.5
#define SROW      1040   // padded f32 row stride for score LDS (1040 % 64 == 16)
#define PROW      1056   // padded f16 row stride for prob LDS (2112B stride, 528 dw, %64==16)
#define ATTN_LDS  (16 * SROW * 4 + 16 * PROW * 2)  // 66560 + 33792 = 100352 B

// ---------- WMMA helpers ----------
__device__ __forceinline__ v8f wmma_f16(v16h a, v16h b, v8f c) {
  // D = A(16x32 f16) * B(32x16 f16) + C(16x16 f32)
  return __builtin_amdgcn_wmma_f32_16x16x32_f16(false, a, false, b, (short)0, c,
                                                false, false);
}

// A fragment: rows m = lane&15 of 16-row block at Abase (row-major, lda halves),
// K chunk [k0, k0+32). Layout: lanes 0-15 hold K k0+0..15,
// lanes 16-31 hold K k0+16..31, 2 halves per VGPR (matches load_b_frag).
__device__ __forceinline__ v16h load_a_frag(const _Float16* Abase, int lda, int k0) {
  int lane = threadIdx.x & 31;
  const _Float16* p = Abase + (lane & 15) * lda + k0 + ((lane >> 4) << 4);
  v8h lo = *(const v8h*)(p);
  v8h hi = *(const v8h*)(p + 8);
  v16h r;
#pragma unroll
  for (int i = 0; i < 8; ++i) { r[i] = lo[i]; r[i + 8] = hi[i]; }
  return r;
}

// B fragment: Bt stored [N][K] (column of B contiguous along K, ldb halves).
// Bbase points at row n0. Lane = column N = n0 + (lane&15); lanes 0-15 hold
// K k0..k0+15, lanes 16-31 hold K k0+16..k0+31 -> one 32B contiguous load.
__device__ __forceinline__ v16h load_b_frag(const _Float16* Bbase, int ldb, int k0) {
  int lane = threadIdx.x & 31;
  const _Float16* p = Bbase + (lane & 15) * ldb + k0 + ((lane >> 4) << 4);
  return *(const v16h*)p;
}

// ---------- LayerNorm (f32 in -> f16 out), one 256-thread block per row ----------
__global__ __launch_bounds__(256)
void ln_kernel(const float* __restrict__ x, const float* __restrict__ g,
               const float* __restrict__ be, _Float16* __restrict__ xn) {
  __shared__ float red[256];
  int row = blockIdx.x;
  int tid = threadIdx.x;
  const float* xr = x + row * DIMC;
  float v[4];
  float s = 0.f;
#pragma unroll
  for (int i = 0; i < 4; ++i) { v[i] = xr[tid + 256 * i]; s += v[i]; }
  red[tid] = s; __syncthreads();
  for (int off = 128; off > 0; off >>= 1) {
    if (tid < off) red[tid] += red[tid + off];
    __syncthreads();
  }
  float mu = red[0] * (1.0f / DIMC);
  __syncthreads();
  float s2 = 0.f;
#pragma unroll
  for (int i = 0; i < 4; ++i) { float d = v[i] - mu; s2 += d * d; }
  red[tid] = s2; __syncthreads();
  for (int off = 128; off > 0; off >>= 1) {
    if (tid < off) red[tid] += red[tid + off];
    __syncthreads();
  }
  float rs = rsqrtf(red[0] * (1.0f / DIMC) + 1e-5f);
#pragma unroll
  for (int i = 0; i < 4; ++i) {
    int c = tid + 256 * i;
    xn[row * DIMC + c] = (_Float16)((v[i] - mu) * rs * g[c] + be[c]);
  }
}

// ---------- f32 -> f16 elementwise ----------
__global__ void cvt_h(const float* __restrict__ in, _Float16* __restrict__ out, int n) {
  int i = blockIdx.x * blockDim.x + threadIdx.x;
  if (i < n) out[i] = (_Float16)in[i];
}

// ---------- transpose f32 [K][N] -> f16 [N][K] ----------
__global__ void transpose_h(const float* __restrict__ in, _Float16* __restrict__ outT,
                            int K, int N) {
  int idx = blockIdx.x * blockDim.x + threadIdx.x;
  if (idx >= K * N) return;
  int k = idx / N, n = idx % N;
  outT[n * K + k] = (_Float16)in[idx];
}

// ---------- templated WMMA GEMM: one wave -> 16x64 strip ----------
// MODE 0: f32 row-major out (final projection -> d_out)
// MODE 1: q epilogue: scale by 1/8, scatter f16 to [b][h][n][64]
// MODE 2: kv epilogue: cols <512 -> k [b][h][j][64]; cols >=512 -> v^T [b][h][64][j]
template <int MODE>
__global__ __launch_bounds__(32)
void gemm_ws(const _Float16* __restrict__ A, const _Float16* __restrict__ Bt,
             void* __restrict__ out0, void* __restrict__ out1,
             int M, int N, int K) {
  int tiles_n = N >> 6;
  int bm = (int)blockIdx.x / tiles_n;
  int bn = (int)blockIdx.x % tiles_n;
  int m0 = bm << 4, n0 = bn << 6;

  v8f acc[4] = {};
  const _Float16* Ab = A + m0 * K;
  for (int k0 = 0; k0 < K; k0 += 32) {
    v16h a = load_a_frag(Ab, K, k0);
#pragma unroll
    for (int t = 0; t < 4; ++t) {
      v16h b = load_b_frag(Bt + (n0 + (t << 4)) * K, K, k0);
      acc[t] = wmma_f16(a, b, acc[t]);
    }
  }

  int lane = threadIdx.x & 31;
  int nl = lane & 15;
  int mbase = (lane < 16) ? 0 : 8;
#pragma unroll
  for (int t = 0; t < 4; ++t) {
#pragma unroll
    for (int v = 0; v < 8; ++v) {
      int m = m0 + mbase + v;
      int n = n0 + (t << 4) + nl;
      float val = acc[t][v];
      if (MODE == 0) {
        ((float*)out0)[m * N + n] = val;
      } else if (MODE == 1) {
        int b = m >> 11, i = m & 2047, h = n >> 6, d = n & 63;
        ((_Float16*)out0)[((((b << 3) + h) << 11) + i) * DHEAD + d] =
            (_Float16)(val * SCALE_Q);
      } else {
        int b = m >> 10, j = m & 1023;
        if (n < INNER) {
          int h = n >> 6, d = n & 63;
          ((_Float16*)out0)[((((b << 3) + h) << 10) + j) * DHEAD + d] = (_Float16)val;
        } else {
          int c = n - INNER, h = c >> 6, d = c & 63;
          ((_Float16*)out1)[((((b << 3) + h) * DHEAD + d) << 10) + j] = (_Float16)val;
        }
      }
    }
  }
}

// ---------- fused attention: one wave per (b,h,16 query rows) ----------
__global__ __launch_bounds__(32)
void attn_kernel(const _Float16* __restrict__ qh,   // [16][2048][64] prescaled
                 const _Float16* __restrict__ kh,   // [16][1024][64]
                 const _Float16* __restrict__ vth,  // [16][64][1024]
                 _Float16* __restrict__ o) {        // [b][2048][512]
  extern __shared__ char smem[];
  float*    S = (float*)smem;                       // 16 x SROW f32 scores
  _Float16* P = (_Float16*)(smem + 16 * SROW * 4);  // 16 x PROW f16 probs

  int bh   = (int)blockIdx.x >> 7;  // 0..15 = b*8+h
  int it   = (int)blockIdx.x & 127;
  int row0 = it << 4;
  int lane = threadIdx.x & 31;
  int nl = lane & 15;
  int mbase = (lane < 16) ? 0 : 8;

  const _Float16* Qb = qh + ((bh << 11) + row0) * DHEAD;
  v16h qa0 = load_a_frag(Qb, DHEAD, 0);
  v16h qa1 = load_a_frag(Qb, DHEAD, 32);

  // Phase 1: S = Q K^T  (B column = K row, contiguous over d)
  for (int jt = 0; jt < 64; ++jt) {
    int j0 = jt << 4;
    const _Float16* Kb = kh + ((bh << 10) + j0) * DHEAD;
    v8f s = {};
    s = wmma_f16(qa0, load_b_frag(Kb, DHEAD, 0), s);
    s = wmma_f16(qa1, load_b_frag(Kb, DHEAD, 32), s);
#pragma unroll
    for (int v = 0; v < 8; ++v)
      S[(mbase + v) * SROW + j0 + nl] = s[v];
  }
  __syncthreads();

  // Phase 2: stable softmax. Two lanes per row (lane r: cols [0,512),
  // lane r+16: cols [512,1024)), combined via wave32 shfl_xor(16).
  // 4-wide vector sweeps over LDS. P is a disjoint LDS region -> no hazards.
  {
    float*    srow = S + nl * SROW;
    _Float16* prow = P + nl * PROW;
    int jbeg = (lane >> 4) << 9;       // 0 or 512
    int jend = jbeg + 512;

    float m = -3.40282347e38f;
    for (int j = jbeg; j < jend; j += 4) {
      v4f xv = *(const v4f*)(srow + j);
      m = fmaxf(m, fmaxf(fmaxf(xv[0], xv[1]), fmaxf(xv[2], xv[3])));
    }
    m = fmaxf(m, __shfl_xor(m, 16));

    float sum = 0.f;
    for (int j = jbeg; j < jend; j += 4) {
      v4f xv = *(const v4f*)(srow + j);
      sum += __expf(xv[0] - m) + __expf(xv[1] - m) +
             __expf(xv[2] - m) + __expf(xv[3] - m);
    }
    sum += __shfl_xor(sum, 16);
    float inv = 1.0f / sum;

    for (int j = jbeg; j < jend; j += 4) {
      v4f xv = *(const v4f*)(srow + j);
      v4h pv;
      pv[0] = (_Float16)(__expf(xv[0] - m) * inv);
      pv[1] = (_Float16)(__expf(xv[1] - m) * inv);
      pv[2] = (_Float16)(__expf(xv[2] - m) * inv);
      pv[3] = (_Float16)(__expf(xv[3] - m) * inv);
      *(v4h*)(prow + j) = pv;
    }
  }
  __syncthreads();

  // Phase 3: O = P @ V  (V^T stored so B column = contiguous over j)
  v8f oacc[4] = {};
  for (int j0 = 0; j0 < NJ; j0 += 32) {
    v16h pa = load_a_frag(P, PROW, j0);
#pragma unroll
    for (int t = 0; t < 4; ++t) {
      v16h b = load_b_frag(vth + ((bh * DHEAD) + (t << 4)) * NJ, NJ, j0);
      oacc[t] = wmma_f16(pa, b, oacc[t]);
    }
  }

  // Phase 4: scatter f16 to [b][n][h*64+d]
  int b = bh >> 3, h = bh & 7;
#pragma unroll
  for (int t = 0; t < 4; ++t) {
#pragma unroll
    for (int v = 0; v < 8; ++v) {
      int m = row0 + mbase + v;
      int d = (h << 6) + (t << 4) + nl;
      o[(((b << 11) + m) * INNER) + d] = (_Float16)oacc[t][v];
    }
  }
}

// ---------- launch ----------
extern "C" void kernel_launch(void* const* d_in, const int* in_sizes, int n_in,
                              void* d_out, int out_size, void* d_ws, size_t ws_size,
                              hipStream_t stream) {
  const float* x     = (const float*)d_in[0];
  const float* media = (const float*)d_in[1];
  const float* g     = (const float*)d_in[2];
  const float* be    = (const float*)d_in[3];
  const float* Wq    = (const float*)d_in[4];
  const float* Wkv   = (const float*)d_in[5];
  const float* Wout  = (const float*)d_in[6];
  float* out = (float*)d_out;

  char* ws = (char*)d_ws;
  _Float16* xn_h   = (_Float16*)(ws + 0);          // 4096x1024  (8 MB)
  _Float16* med_h  = (_Float16*)(ws + 8388608);    // 2048x1024  (4 MB)
  _Float16* Wq_t   = (_Float16*)(ws + 12582912);   // 512x1024   (1 MB)
  _Float16* Wkv_t  = (_Float16*)(ws + 13631488);   // 1024x1024  (2 MB)
  _Float16* Wout_t = (_Float16*)(ws + 15728640);   // 1024x512   (1 MB)
  _Float16* q_h    = (_Float16*)(ws + 16777216);   // 16x2048x64 (4 MB)
  _Float16* k_h    = (_Float16*)(ws + 20971520);   // 16x1024x64 (2 MB)
  _Float16* vt_h   = (_Float16*)(ws + 23068672);   // 16x64x1024 (2 MB)
  _Float16* ao_h   = (_Float16*)(ws + 25165824);   // 4096x512   (4 MB)

  // Preprocess: LN + f16 casts + weight transposes
  ln_kernel<<<NB * NQ, 256, 0, stream>>>(x, g, be, xn_h);
  cvt_h<<<(NB * NJ * DIMC) / 256, 256, 0, stream>>>(media, med_h, NB * NJ * DIMC);
  transpose_h<<<(DIMC * INNER) / 256, 256, 0, stream>>>(Wq, Wq_t, DIMC, INNER);
  transpose_h<<<(DIMC * DIMC) / 256, 256, 0, stream>>>(Wkv, Wkv_t, DIMC, 2 * INNER);
  transpose_h<<<(INNER * DIMC) / 256, 256, 0, stream>>>(Wout, Wout_t, INNER, DIMC);

  // q = xn @ Wq (scaled, scattered per-head)
  gemm_ws<1><<<(NB * NQ / 16) * (INNER / 64), 32, 0, stream>>>(
      xn_h, Wq_t, (void*)q_h, nullptr, NB * NQ, INNER, DIMC);
  // kv = media @ Wkv (k row-major per head, v transposed per head)
  gemm_ws<2><<<(NB * NJ / 16) * (2 * INNER / 64), 32, 0, stream>>>(
      med_h, Wkv_t, (void*)k_h, (void*)vt_h, NB * NJ, 2 * INNER, DIMC);
  // fused softmax(QK^T)V
  attn_kernel<<<16 * (NQ / 16), 32, ATTN_LDS, stream>>>(q_h, k_h, vt_h, ao_h);
  // final projection -> f32 output
  gemm_ws<0><<<(NB * NQ / 16) * (DIMC / 64), 32, 0, stream>>>(
      ao_h, Wout_t, (void*)out, nullptr, NB * NQ, DIMC, INNER);
}
